// Block_75952201662853
// MI455X (gfx1250) — matrix operand order repaired and checked by reading, not compiled, not run on hardware
//
#include <hip/hip_runtime.h>
#include <cstdint>
#include <cstddef>

// ---------------------------------------------------------------------------
// Types for CDNA5 WMMA (gfx1250, wave32)
// ---------------------------------------------------------------------------
typedef __attribute__((ext_vector_type(16))) __bf16 v16bf;
typedef __attribute__((ext_vector_type(8)))  float  v8f;

union Frag {
    v16bf v;
    float4 f2[2];   // 32 bytes = 8 VGPRs per lane
};

__device__ __forceinline__ __bf16 f2bf(float f) {
    unsigned u = __builtin_bit_cast(unsigned, f);
    u = (u + 0x7FFFu + ((u >> 16) & 1u)) >> 16;   // round-to-nearest-even
    unsigned short s = (unsigned short)u;
    return __builtin_bit_cast(__bf16, s);
}

__device__ __forceinline__ v16bf ld_fragB_contig(const __bf16* p) {
    Frag f;
    f.f2[0] = *(const float4*)(p);
    f.f2[1] = *(const float4*)(p + 8);
    return f.v;
}

__device__ __forceinline__ v16bf ld_fragA_rowmajor(const __bf16* p) {
    Frag f;
    f.f2[0] = *(const float4*)(p);
    f.f2[1] = *(const float4*)(p + 8);
    return f.v;
}

#define WMMA_BF16(A, B, C) \
    __builtin_amdgcn_wmma_f32_16x16x32_bf16(false, (A), false, (B), (short)0, (C), false, false)

// ---------------------------------------------------------------------------
// CDNA5 async global->LDS path (ASYNCcnt-tracked), with sync fallback.
// Builtin signature (from compiler diagnostic): param1 is a vector_size(16)
// int pointer in the global (AS1) address space; param2 the LDS (AS3) one.
// ---------------------------------------------------------------------------
#if defined(__HIP_DEVICE_COMPILE__) && __has_builtin(__builtin_amdgcn_global_load_async_to_lds_b128)
#define USE_ASYNC_LDS 1
#else
#define USE_ASYNC_LDS 0
#endif

#if USE_ASYNC_LDS
typedef int v4i __attribute__((__vector_size__(4 * sizeof(int))));
typedef __attribute__((address_space(1))) v4i gv4i_t;   // global
typedef __attribute__((address_space(3))) v4i lv4i_t;   // LDS
__device__ __forceinline__ gv4i_t* to_global(const void* p) {
    return (gv4i_t*)(uintptr_t)p;
}
__device__ __forceinline__ lv4i_t* to_lds(void* p) {
    return (lv4i_t*)(uintptr_t)p;   // generic LDS ptr low 32 bits == LDS offset
}
#endif

template <int N>
__device__ __forceinline__ void wait_async() {
#if USE_ASYNC_LDS
#if __has_builtin(__builtin_amdgcn_s_wait_asynccnt)
    __builtin_amdgcn_s_wait_asynccnt(N);
#else
    asm volatile("s_wait_asynccnt %0" ::"i"(N) : "memory");
#endif
#endif
}

// Stage 32 bytes of A and 32 bytes of B per thread (global -> LDS).
__device__ __forceinline__ void stage_pair(const __bf16* gA, const __bf16* gB,
                                           __bf16* lA, __bf16* lB) {
#if USE_ASYNC_LDS
    __builtin_amdgcn_global_load_async_to_lds_b128(to_global(gA),     to_lds(lA),     0, 0);
    __builtin_amdgcn_global_load_async_to_lds_b128(to_global(gA + 8), to_lds(lA + 8), 0, 0);
    __builtin_amdgcn_global_load_async_to_lds_b128(to_global(gB),     to_lds(lB),     0, 0);
    __builtin_amdgcn_global_load_async_to_lds_b128(to_global(gB + 8), to_lds(lB + 8), 0, 0);
#else
    float4 a0 = *(const float4*)(gA);
    float4 a1 = *(const float4*)(gA + 8);
    float4 b0 = *(const float4*)(gB);
    float4 b1 = *(const float4*)(gB + 8);
    *(float4*)(lA)     = a0;
    *(float4*)(lA + 8) = a1;
    *(float4*)(lB)     = b0;
    *(float4*)(lB + 8) = b1;
#endif
}

// ---------------------------------------------------------------------------
// Constants
// ---------------------------------------------------------------------------
static constexpr int Bb = 4;
static constexpr int Tt = 2048;
static constexpr int Cc = 1024;
static constexpr int Hh = 16;
static constexpr int Dd = 64;
static constexpr int Ff = 4096;
static constexpr int Mrows = Bb * Tt;     // 8192

// ---------------------------------------------------------------------------
// One-time weight convert + transpose: W[K][N] fp32 -> Wt[N][K] bf16
// ---------------------------------------------------------------------------
__global__ void __launch_bounds__(256)
wt_convert_kernel(const float* __restrict__ W, __bf16* __restrict__ Wt, int K, int N) {
    __shared__ float tile[32][33];
    const int kb = blockIdx.y * 32, nb = blockIdx.x * 32;
    const int tx = threadIdx.x & 31;
    const int ty = threadIdx.x >> 5;   // 0..7
#pragma unroll
    for (int i = 0; i < 32; i += 8)
        tile[ty + i][tx] = W[(size_t)(kb + ty + i) * N + nb + tx];
    __syncthreads();
#pragma unroll
    for (int i = 0; i < 32; i += 8)
        Wt[(size_t)(nb + ty + i) * K + kb + tx] = f2bf(tile[tx][ty + i]);
}

// ---------------------------------------------------------------------------
// LayerNorm: one 256-thread block per row of C=1024, bf16 output
// ---------------------------------------------------------------------------
__global__ void __launch_bounds__(256)
ln_kernel(const float* __restrict__ x, const float* __restrict__ g,
          const float* __restrict__ b, __bf16* __restrict__ out) {
    const int row = blockIdx.x;
    const int lane = threadIdx.x & 31;
    const int wid  = threadIdx.x >> 5;
    const float* xr = x + (size_t)row * Cc;
    const int c0 = threadIdx.x * 4;

    float xv[4];
    float s = 0.f, s2 = 0.f;
#pragma unroll
    for (int j = 0; j < 4; ++j) {
        xv[j] = xr[c0 + j];
        s  += xv[j];
        s2 += xv[j] * xv[j];
    }
#pragma unroll
    for (int m = 1; m < 32; m <<= 1) {
        s  += __shfl_xor(s,  m, 32);
        s2 += __shfl_xor(s2, m, 32);
    }
    __shared__ float red[2][8];
    if (lane == 0) { red[0][wid] = s; red[1][wid] = s2; }
    __syncthreads();
    float ts = 0.f, ts2 = 0.f;
#pragma unroll
    for (int i = 0; i < 8; ++i) { ts += red[0][i]; ts2 += red[1][i]; }
    const float mu  = ts * (1.0f / Cc);
    const float var = ts2 * (1.0f / Cc) - mu * mu;
    const float r   = rsqrtf(var + 1e-5f);
#pragma unroll
    for (int j = 0; j < 4; ++j) {
        const int c = c0 + j;
        out[(size_t)row * Cc + c] = f2bf((xv[j] - mu) * r * g[c] + b[c]);
    }
}

// ---------------------------------------------------------------------------
// Tiled bf16 WMMA GEMM:  C[M,N] = A[M,K](bf16) * Wt[N,K](bf16, pre-transposed)
// Block tile 128x128, BK=32, double-buffered LDS with async global->LDS DMA.
// ---------------------------------------------------------------------------
template <bool RELU, bool RESID, bool OUTBF>
__global__ void __launch_bounds__(256)
gemm_bf16_kernel(const __bf16* __restrict__ A, const __bf16* __restrict__ Wt,
                 const float* __restrict__ bias, const float* __restrict__ resid,
                 float* __restrict__ Cf, __bf16* __restrict__ Cb,
                 int M, int N, int Kd) {
    __shared__ __bf16 As[2][128 * 40];   // [row][k], padded stride 40
    __shared__ __bf16 Bt[2][128 * 40];   // [n][k],  padded stride 40

    const int lane = threadIdx.x & 31;
    const int wid  = threadIdx.x >> 5;
    const int hi   = lane >> 4;
    const int ln   = lane & 15;
    const int rowBase = blockIdx.y * 128;
    const int colBase = blockIdx.x * 128;
    const int waveM = wid & 1;
    const int waveN = wid >> 1;

    const int r    = threadIdx.x >> 1;     // 0..127
    const int half = threadIdx.x & 1;      // 0..1
    const int lOff = r * 40 + half * 16;
    const __bf16* gA = A  + (size_t)(rowBase + r) * Kd + half * 16;
    const __bf16* gB = Wt + (size_t)(colBase + r) * Kd + half * 16;

    const v8f vzero = {0.f, 0.f, 0.f, 0.f, 0.f, 0.f, 0.f, 0.f};
    v8f acc[4][2];
#pragma unroll
    for (int mt = 0; mt < 4; ++mt)
#pragma unroll
        for (int nt = 0; nt < 2; ++nt) acc[mt][nt] = vzero;

    // prologue: stage K-tile 0 into buffer 0
    stage_pair(gA, gB, &As[0][lOff], &Bt[0][lOff]);

    int cur = 0;
    for (int kb = 0; kb < Kd; kb += 32) {
        const bool hasNext = (kb + 32) < Kd;
        if (hasNext)   // overlap next tile's DMA with this tile's WMMAs
            stage_pair(gA + kb + 32, gB + kb + 32, &As[cur ^ 1][lOff], &Bt[cur ^ 1][lOff]);
        if (hasNext) wait_async<4>();   // drain the 4 oldest (current buffer) DMAs
        else         wait_async<0>();
        __syncthreads();

        Frag bfr[2];
#pragma unroll
        for (int nt = 0; nt < 2; ++nt) {
            const __bf16* bp = &Bt[cur][(waveN * 32 + nt * 16 + ln) * 40 + hi * 16];
            bfr[nt].f2[0] = *(const float4*)(bp);
            bfr[nt].f2[1] = *(const float4*)(bp + 8);
        }
#pragma unroll
        for (int mt = 0; mt < 4; ++mt) {
            const __bf16* apl = &As[cur][(waveM * 64 + mt * 16 + ln) * 40 + hi * 16];
            Frag af;
            af.f2[0] = *(const float4*)(apl);
            af.f2[1] = *(const float4*)(apl + 8);
#pragma unroll
            for (int nt = 0; nt < 2; ++nt)
                acc[mt][nt] = WMMA_BF16(af.v, bfr[nt].v, acc[mt][nt]);
        }
        __syncthreads();   // all waves done with buffer `cur` before it is refilled
        cur ^= 1;
    }

    // epilogue: D-layout is (M = e + 8*hi, N = ln) per 16x16 tile
#pragma unroll
    for (int mt = 0; mt < 4; ++mt) {
#pragma unroll
        for (int nt = 0; nt < 2; ++nt) {
#pragma unroll
            for (int e = 0; e < 8; ++e) {
                const int row = rowBase + waveM * 64 + mt * 16 + e + 8 * hi;
                const int col = colBase + waveN * 32 + nt * 16 + ln;
                float v = acc[mt][nt][e];
                if (bias) v += bias[col];
                if (RELU) v = fmaxf(v, 0.f);
                if (RESID) v += resid[(size_t)row * N + col];
                if (OUTBF) Cb[(size_t)row * N + col] = f2bf(v);
                else       Cf[(size_t)row * N + col] = v;
            }
        }
    }
    (void)M;
}

// ---------------------------------------------------------------------------
// Causal flash attention.  Grid: (T/64, H, B), block = 128 threads (4 waves).
// ---------------------------------------------------------------------------
__global__ void __launch_bounds__(128)
attn_kernel(const __bf16* __restrict__ Q, const __bf16* __restrict__ K,
            const __bf16* __restrict__ V, __bf16* __restrict__ O) {
    const int lane = threadIdx.x & 31;
    const int wid  = threadIdx.x >> 5;
    const int hi   = lane >> 4;
    const int ln   = lane & 15;
    const int h  = blockIdx.y;
    const int bb = blockIdx.z;
    const int q0block = blockIdx.x * 64;
    const int q0 = q0block + wid * 16;

    __shared__ __bf16 vt[64 * 40];        // V^T stage: [d][key], stride 40
    __shared__ __bf16 sP[4][16 * 32];     // per-wave P tile [m][key]

    const __bf16* qp = Q + (size_t)(bb * Tt + q0 + ln) * Cc + h * Dd;
    Frag aq0, aq1;
    aq0.f2[0] = *(const float4*)(qp + hi * 16);
    aq0.f2[1] = *(const float4*)(qp + hi * 16 + 8);
    aq1.f2[0] = *(const float4*)(qp + 32 + hi * 16);
    aq1.f2[1] = *(const float4*)(qp + 32 + hi * 16 + 8);

    const v8f vzero = {0.f, 0.f, 0.f, 0.f, 0.f, 0.f, 0.f, 0.f};
    v8f o[4];
#pragma unroll
    for (int i = 0; i < 4; ++i) o[i] = vzero;
    float mrow[8], lrow[8];               // row = q0 + e + 8*hi
#pragma unroll
    for (int e = 0; e < 8; ++e) { mrow[e] = -3.0e38f; lrow[e] = 0.f; }

    const int myEnd    = q0 + 16;
    const int blockEnd = q0block + 64;

    for (int kb = 0; kb < blockEnd; kb += 32) {
        __syncthreads();
        {   // cooperative V^T stage: 32 keys x 64 d
            const int kk   = threadIdx.x >> 2;         // 0..31
            const int dseg = (threadIdx.x & 3) * 16;   // 0,16,32,48
            const __bf16* vp = V + (size_t)(bb * Tt + kb + kk) * Cc + h * Dd + dseg;
            float4 u0 = *(const float4*)(vp);
            float4 u1 = *(const float4*)(vp + 8);
            const __bf16* e0 = (const __bf16*)&u0;
            const __bf16* e1 = (const __bf16*)&u1;
#pragma unroll
            for (int j = 0; j < 8; ++j) {
                vt[(dseg + j)     * 40 + kk] = e0[j];
                vt[(dseg + 8 + j) * 40 + kk] = e1[j];
            }
        }
        __syncthreads();
        if (kb >= myEnd) continue;   // keep barriers uniform across the block

        // scores: two 16-key tiles, B-frags straight from global K
        v8f s[2];
#pragma unroll
        for (int nt = 0; nt < 2; ++nt) {
            const __bf16* kp = K + (size_t)(bb * Tt + kb + nt * 16 + ln) * Cc + h * Dd + hi * 16;
            v8f a = vzero;
            a = WMMA_BF16(aq0.v, ld_fragB_contig(kp),      a);
            a = WMMA_BF16(aq1.v, ld_fragB_contig(kp + 32), a);
            s[nt] = a;
        }

        // online softmax
        float nm[8];
#pragma unroll
        for (int e = 0; e < 8; ++e) {
            const int row = q0 + e + 8 * hi;
            float v0 = s[0][e] * 0.125f;
            float v1 = s[1][e] * 0.125f;
            if (kb + ln      > row) v0 = -3.0e38f;
            if (kb + 16 + ln > row) v1 = -3.0e38f;
            s[0][e] = v0; s[1][e] = v1;
            float mx = fmaxf(v0, v1);
#pragma unroll
            for (int m = 1; m < 16; m <<= 1) mx = fmaxf(mx, __shfl_xor(mx, m, 32));
            nm[e] = fmaxf(mrow[e], mx);
        }
#pragma unroll
        for (int e = 0; e < 8; ++e) {
            const float p0 = __expf(s[0][e] - nm[e]);
            const float p1 = __expf(s[1][e] - nm[e]);
            const float alpha = __expf(mrow[e] - nm[e]);
            float rs = p0 + p1;
#pragma unroll
            for (int m = 1; m < 16; m <<= 1) rs += __shfl_xor(rs, m, 32);
            lrow[e] = lrow[e] * alpha + rs;
            mrow[e] = nm[e];
            o[0][e] *= alpha; o[1][e] *= alpha; o[2][e] *= alpha; o[3][e] *= alpha;
            const int m16 = e + 8 * hi;
            sP[wid][m16 * 32 + ln]      = f2bf(p0);
            sP[wid][m16 * 32 + 16 + ln] = f2bf(p1);
        }

        // P @ V
        const __bf16* pp = &sP[wid][ln * 32 + hi * 16];
        const v16bf ap = ld_fragA_rowmajor(pp);
#pragma unroll
        for (int nt2 = 0; nt2 < 4; ++nt2) {
            const __bf16* vpp = &vt[(nt2 * 16 + ln) * 40 + hi * 16];
            o[nt2] = WMMA_BF16(ap, ld_fragB_contig(vpp), o[nt2]);
        }
    }

    // finalize
#pragma unroll
    for (int nt2 = 0; nt2 < 4; ++nt2) {
#pragma unroll
        for (int e = 0; e < 8; ++e) {
            const int row = q0 + e + 8 * hi;
            const int d   = nt2 * 16 + ln;
            O[(size_t)(bb * Tt + row) * Cc + h * Dd + d] = f2bf(o[nt2][e] * (1.0f / lrow[e]));
        }
    }
}

// ---------------------------------------------------------------------------
// Launcher
// ---------------------------------------------------------------------------
extern "C" void kernel_launch(void* const* d_in, const int* in_sizes, int n_in,
                              void* d_out, int out_size, void* d_ws, size_t ws_size,
                              hipStream_t stream) {
    const float* x    = (const float*)d_in[0];
    const float* Wq   = (const float*)d_in[1];
    const float* Wk   = (const float*)d_in[2];
    const float* Wv   = (const float*)d_in[3];
    const float* Wo   = (const float*)d_in[4];
    const float* bo   = (const float*)d_in[5];
    const float* ln1g = (const float*)d_in[6];
    const float* ln1b = (const float*)d_in[7];
    const float* ln2g = (const float*)d_in[8];
    const float* ln2b = (const float*)d_in[9];
    const float* W1   = (const float*)d_in[10];
    const float* b1   = (const float*)d_in[11];
    const float* W2   = (const float*)d_in[12];
    const float* b2   = (const float*)d_in[13];
    float* out = (float*)d_out;

    // workspace carve (~216 MiB)
    char* ws = (char*)d_ws;
    size_t off = 0;
    auto carve = [&](size_t bytes) {
        void* p = ws + off;
        off += (bytes + 255) & ~(size_t)255;
        return p;
    };
    const size_t nMC = (size_t)Mrows * Cc;
    __bf16* hb   = (__bf16*)carve(nMC * 2);                 // LN1 out
    __bf16* qb   = (__bf16*)carve(nMC * 2);
    __bf16* kb   = (__bf16*)carve(nMC * 2);
    __bf16* vb   = (__bf16*)carve(nMC * 2);
    __bf16* ob   = (__bf16*)carve(nMC * 2);                 // attn out
    float*  x1   = (float*) carve(nMC * 4);                 // x + attn proj
    __bf16* h2b  = (__bf16*)carve(nMC * 2);                 // LN2 out
    __bf16* actb = (__bf16*)carve((size_t)Mrows * Ff * 2);  // ReLU(h2 @ W1 + b1)
    __bf16* Wqt  = (__bf16*)carve((size_t)Cc * Cc * 2);     // transposed bf16 weights
    __bf16* Wkt  = (__bf16*)carve((size_t)Cc * Cc * 2);
    __bf16* Wvt  = (__bf16*)carve((size_t)Cc * Cc * 2);
    __bf16* Wot  = (__bf16*)carve((size_t)Cc * Cc * 2);
    __bf16* W1t  = (__bf16*)carve((size_t)Cc * Ff * 2);     // [N=4096][K=1024]
    __bf16* W2t  = (__bf16*)carve((size_t)Cc * Ff * 2);     // [N=1024][K=4096]
    (void)ws_size; (void)n_in; (void)in_sizes; (void)out_size;

    const dim3 blk256(256), blk128(128);
    const dim3 gC(Cc / 128, Mrows / 128);   // N=1024 GEMMs
    const dim3 gF(Ff / 128, Mrows / 128);   // N=4096 GEMM

    // one-time (per launch) weight convert + transpose: bandwidth-trivial
    wt_convert_kernel<<<dim3(Cc / 32, Cc / 32), blk256, 0, stream>>>(Wq, Wqt, Cc, Cc);
    wt_convert_kernel<<<dim3(Cc / 32, Cc / 32), blk256, 0, stream>>>(Wk, Wkt, Cc, Cc);
    wt_convert_kernel<<<dim3(Cc / 32, Cc / 32), blk256, 0, stream>>>(Wv, Wvt, Cc, Cc);
    wt_convert_kernel<<<dim3(Cc / 32, Cc / 32), blk256, 0, stream>>>(Wo, Wot, Cc, Cc);
    wt_convert_kernel<<<dim3(Ff / 32, Cc / 32), blk256, 0, stream>>>(W1, W1t, Cc, Ff);
    wt_convert_kernel<<<dim3(Cc / 32, Ff / 32), blk256, 0, stream>>>(W2, W2t, Ff, Cc);

    // LN1
    ln_kernel<<<Mrows, blk256, 0, stream>>>(x, ln1g, ln1b, hb);
    // QKV projections
    gemm_bf16_kernel<false, false, true><<<gC, blk256, 0, stream>>>(
        hb, Wqt, nullptr, nullptr, nullptr, qb, Mrows, Cc, Cc);
    gemm_bf16_kernel<false, false, true><<<gC, blk256, 0, stream>>>(
        hb, Wkt, nullptr, nullptr, nullptr, kb, Mrows, Cc, Cc);
    gemm_bf16_kernel<false, false, true><<<gC, blk256, 0, stream>>>(
        hb, Wvt, nullptr, nullptr, nullptr, vb, Mrows, Cc, Cc);
    // causal flash attention
    attn_kernel<<<dim3(Tt / 64, Hh, Bb), blk128, 0, stream>>>(qb, kb, vb, ob);
    // output projection + bias + residual -> x1 (fp32)
    gemm_bf16_kernel<false, true, false><<<gC, blk256, 0, stream>>>(
        ob, Wot, bo, x, x1, nullptr, Mrows, Cc, Cc);
    // LN2
    ln_kernel<<<Mrows, blk256, 0, stream>>>(x1, ln2g, ln2b, h2b);
    // FFN up + ReLU (bf16 out)
    gemm_bf16_kernel<true, false, true><<<gF, blk256, 0, stream>>>(
        h2b, W1t, b1, nullptr, nullptr, actb, Mrows, Ff, Cc);
    // FFN down + bias + residual -> d_out (fp32)
    gemm_bf16_kernel<false, true, false><<<gC, blk256, 0, stream>>>(
        actb, W2t, b2, x1, out, nullptr, Mrows, Cc, Ff);
}
